// JumpODE_73770358276751
// MI455X (gfx1250) — matrix operand order repaired and checked by reading, not compiled, not run on hardware
//
#include <hip/hip_runtime.h>
#include <hip/hip_bf16.h>

// Problem constants
#define BB 512
#define TT 512
#define CC 128
#define HH 128

typedef __bf16 bf16;
typedef __attribute__((ext_vector_type(16))) __bf16 v16bf;
typedef __attribute__((ext_vector_type(8)))  __bf16 v8bf;
typedef __attribute__((ext_vector_type(8)))  float  v8f;

// Global (address_space(1)) flavors so weight fetches lower to
// global_load_b128 (saddr + voffset) instead of flat_load_b128.
typedef __attribute__((address_space(1))) const bf16  GBF;
typedef __attribute__((address_space(1))) const v16bf GV16BF;

static __device__ inline unsigned short f2bfbits(float f) {
  unsigned u = __builtin_bit_cast(unsigned, f);
  u += 0x7FFFu + ((u >> 16) & 1u);   // round-to-nearest-even
  return (unsigned short)(u >> 16);
}
static __device__ inline bf16 f2bf(float f) {
  unsigned short s = f2bfbits(f);
  return __builtin_bit_cast(bf16, s);
}

static __device__ inline v8f vzero8f() {
  v8f z = {0.f, 0.f, 0.f, 0.f, 0.f, 0.f, 0.f, 0.f};
  return z;
}

// Launder a uniform pointer through an empty asm so the optimizer cannot
// prove loads through it loop-invariant (prevents LICM hoisting every
// weight fragment out of the 512-step time loop -> scratch spills).
// Result stays in the GLOBAL address space.
static __device__ inline GBF* opaque_g(const bf16* p) {
  unsigned long long u = (unsigned long long)p;
  asm volatile("" : "+s"(u));
  return (GBF*)u;
}

// A-fragment (16x32 bf16) from LDS. Contiguous k order (as in B): lane = (hlf,m),
// element e -> K = ko + 16*hlf + e           (e < 16)
// Two contiguous 16B chunks -> two ds_load_b128.
static __device__ inline v16bf load_a(const bf16* row, int ko, int hlf) {
  v8bf lo = *(const v8bf*)(row + ko + 16 * hlf);
  v8bf hi = *(const v8bf*)(row + ko + 16 * hlf + 8);
  return __builtin_shufflevector(lo, hi, 0, 1, 2, 3, 4, 5, 6, 7,
                                         8, 9, 10, 11, 12, 13, 14, 15);
}

// B-fragment (32x16 bf16) from global weight row (row-major W[n][K]):
// lane = (hlf, n), element e -> K = e within 16 contiguous bf16 at eoff.
// base is uniform (SGPR pair), eoff is a 32-bit lane offset -> saddr form.
static __device__ inline v16bf load_bg(GBF* base, int eoff) {
  return *(GV16BF*)(base + eoff);
}

static __device__ inline v8f wmma_bf(v16bf a, v16bf b, v8f c) {
  return __builtin_amdgcn_wmma_f32_16x16x32_bf16(false, a, false, b,
                                                 (short)0, c, false, false);
}

static __device__ inline void wave_ln_stats(float s, float q, float& mean, float& rstd) {
#pragma unroll
  for (int o = 1; o < 32; o <<= 1) {
    s += __shfl_xor(s, o, 32);
    q += __shfl_xor(q, o, 32);
  }
  mean = s * (1.0f / 128.0f);
  float var = q * (1.0f / 128.0f) - mean * mean;
  rstd = rsqrtf(var + 1e-5f);
}

// ---------------- weight prep: fp32 -> bf16, packed into d_ws ---------------
// ws layout (bf16 elements):
//   [0, 262144)          W1  row-major [512][512]
//   [262144, 327680)     W2  row-major [128][512]
//   [327680, 425984)     Wg  [384][256] : Wg[j][0:128]=W_ih[j], [128:256]=W_hh[j]
__global__ void jumpode_prep_weights(const float* __restrict__ W1,
                                     const float* __restrict__ W2,
                                     const float* __restrict__ Wih,
                                     const float* __restrict__ Whh,
                                     unsigned short* __restrict__ ws) {
  int i = blockIdx.x * blockDim.x + threadIdx.x;
  if (i < 262144) {
    ws[i] = f2bfbits(W1[i]);
  } else if (i < 262144 + 65536) {
    ws[i] = f2bfbits(W2[i - 262144]);
  } else if (i < 262144 + 65536 + 98304) {
    int j = i - (262144 + 65536);
    int row = j >> 8;
    int col = j & 255;
    float v = (col < 128) ? Wih[row * 128 + col] : Whh[row * 128 + (col - 128)];
    ws[i] = f2bfbits(v);
  }
}

// ------------------------- persistent scan kernel ---------------------------
// 32 blocks x 512 threads (16 wave32). Block owns batch rows [16*bid, 16*bid+16).
__global__ __launch_bounds__(512)
void jumpode_scan(const float* __restrict__ x, const float* __restrict__ ts,
                  const float* __restrict__ b1, const float* __restrict__ b2,
                  const float* __restrict__ b_ih, const float* __restrict__ b_hh,
                  const float* __restrict__ g_ode, const float* __restrict__ beta_ode,
                  const float* __restrict__ g_gru, const float* __restrict__ beta_gru,
                  const bf16* __restrict__ W1b, const bf16* __restrict__ W2b,
                  const bf16* __restrict__ Wgb, float* __restrict__ out) {
  // f32 scratch: base[16][512] overlaid with GRU gates (disjoint live ranges)
  __shared__ float fbuf[16][512];
#define BASE_S(m, n) fbuf[m][n]
#define GG_S(m, n)   fbuf[m][n]            /* n in [0,256): r,z pre-gates  */
#define INB_S(m, c)  fbuf[m][256 + (c)]    /* i_n */
#define HNB_S(m, c)  fbuf[m][384 + (c)]    /* h_n */
  __shared__ float hf[16][128];    // persistent hidden state (f32 master)
  __shared__ float h1f[16][128];   // post-ODE-LN hidden
  __shared__ float kcur[16][128];  // current RK4 slope
  __shared__ float ksum[16][128];  // weighted RK4 slope sum
  // bf16 A-operand staging; strides chosen so (stride mod 64 dwords)==4 -> no bank conflicts
  __shared__ __align__(16) bf16 axls[16][392];  // [xi | xl | sl]  (K=384)
  __shared__ __align__(16) bf16 hkbf[16][136];  // RK4 stage input (K=128)
  __shared__ __align__(16) bf16 sbf[16][520];   // silu(G1)        (K=512)
  __shared__ __align__(16) bf16 gruA[16][264];  // [xi | h1]       (K=256)
  __shared__ float dts_s[16];
  // preloaded parameter vectors
  __shared__ float b1s[512], b2s[128], bsum[256], binn[128], bhnn[128];
  __shared__ float gos[128], bos[128], ggs[128], bgs[128];

  const int tid = threadIdx.x;
  const int wv  = tid >> 5;   // wave id 0..15
  const int lane = tid & 31;
  const int hlf = lane >> 4;  // K-half for WMMA fragments
  const int l16 = lane & 15;  // matrix row (A) / column (B,C)
  const int b0 = blockIdx.x * 16;

  // ---- one-time preload ----
  if (tid < 512) b1s[tid] = b1[tid];
  if (tid < 256) bsum[tid] = b_ih[tid] + b_hh[tid];
  if (tid < 128) {
    b2s[tid]  = b2[tid];
    binn[tid] = b_ih[256 + tid];
    bhnn[tid] = b_hh[256 + tid];
    gos[tid] = g_ode[tid];  bos[tid] = beta_ode[tid];
    ggs[tid] = g_gru[tid];  bgs[tid] = beta_gru[tid];
  }
  {
    const int m = wv, c = lane * 4;
#pragma unroll
    for (int j = 0; j < 4; ++j) hf[m][c + j] = 0.0f;
  }
  __syncthreads();

  const float coef[4] = {0.f, 0.5f, 0.5f, 1.f};
  const float wgt[4]  = {1.f, 2.f, 2.f, 1.f};

#pragma unroll 1
  for (int t = 0; t < TT; ++t) {
    // ---- stage 1: per-row dt ----
    if (tid < 16) {
      float tt = ts[(size_t)(b0 + tid) * TT + t];
      dts_s[tid] = (t > 0) ? (tt - ts[(size_t)(b0 + tid) * TT + t - 1]) : 1.0f;
    }
    __syncthreads();

    // ---- stage 2: load xi/xl, build [xi|xl|sl] bf16 and gruA[:,0:128]=xi ----
    {
      const int m = wv, c = lane * 4;
      const float* xr = x + ((size_t)(b0 + m) * TT + t) * CC + c;
      float4 xi4 = *(const float4*)xr;
      float4 xl4 = (t > 0) ? *(const float4*)(xr - CC) : xi4;
      const float dinv = 1.0f / dts_s[m];
      float xi_[4] = {xi4.x, xi4.y, xi4.z, xi4.w};
      float xl_[4] = {xl4.x, xl4.y, xl4.z, xl4.w};
#pragma unroll
      for (int j = 0; j < 4; ++j) {
        float sl = (xi_[j] - xl_[j]) * dinv;
        axls[m][c + j]       = f2bf(xi_[j]);
        axls[m][128 + c + j] = f2bf(xl_[j]);
        axls[m][256 + c + j] = f2bf(sl);
        gruA[m][c + j]       = f2bf(xi_[j]);
      }
    }
    __syncthreads();

    // ---- stage 3: base = [xi|xl|sl] @ W1[:,128:512]^T + b1   (M16,N512,K384)
    {
      const int n0 = wv * 32 + l16;
      const int n1 = n0 + 16;
      GBF* W1o = opaque_g(W1b);            // keep loads inside the t loop
      v8f acc0 = vzero8f(), acc1 = vzero8f();
      const int e0 = n0 * 512 + 128 + 16 * hlf;
      const int e1 = n1 * 512 + 128 + 16 * hlf;
#pragma unroll 3
      for (int kt = 0; kt < 12; ++kt) {
        v16bf a = load_a(&axls[l16][0], kt * 32, hlf);
        acc0 = wmma_bf(a, load_bg(W1o, e0 + kt * 32), acc0);
        acc1 = wmma_bf(a, load_bg(W1o, e1 + kt * 32), acc1);
      }
#pragma unroll
      for (int r = 0; r < 8; ++r) {
        int m = r + 8 * hlf;
        BASE_S(m, n0) = acc0[r] + b1s[n0];
        BASE_S(m, n1) = acc1[r] + b1s[n1];
      }
    }

    // ---- stage 4: RK4 evals ----
#pragma unroll 1
    for (int rk = 0; rk < 4; ++rk) {
      {  // 4a: hk = h + coef*d*k_prev  -> bf16
        const int m = wv, c = lane * 4;
        const float cf = coef[rk] * dts_s[m];
#pragma unroll
        for (int j = 0; j < 4; ++j) {
          float hv = hf[m][c + j];
          if (rk > 0) hv += cf * kcur[m][c + j];
          hkbf[m][c + j] = f2bf(hv);
        }
      }
      __syncthreads();
      {  // 4b: G1 = hk @ W1[:,0:128]^T + base; silu -> sbf  (M16,N512,K128)
        const int n0 = wv * 32 + l16;
        const int n1 = n0 + 16;
        GBF* W1o = opaque_g(W1b);          // re-opaque per rk: no cross-rk hoist
        v8f acc0 = vzero8f(), acc1 = vzero8f();
        const int e0 = n0 * 512 + 16 * hlf;
        const int e1 = n1 * 512 + 16 * hlf;
#pragma unroll 2
        for (int kt = 0; kt < 4; ++kt) {
          v16bf a = load_a(&hkbf[l16][0], kt * 32, hlf);
          acc0 = wmma_bf(a, load_bg(W1o, e0 + kt * 32), acc0);
          acc1 = wmma_bf(a, load_bg(W1o, e1 + kt * 32), acc1);
        }
#pragma unroll
        for (int r = 0; r < 8; ++r) {
          int m = r + 8 * hlf;
          float s0 = acc0[r] + BASE_S(m, n0);
          float s1 = acc1[r] + BASE_S(m, n1);
          s0 = s0 / (1.0f + __expf(-s0));
          s1 = s1 / (1.0f + __expf(-s1));
          sbf[m][n0] = f2bf(s0);
          sbf[m][n1] = f2bf(s1);
        }
      }
      __syncthreads();
      if (wv < 8) {  // 4c: k = silu(G1) @ W2^T + b2   (M16,N128,K512)
        const int j0 = wv * 16 + l16;
        GBF* W2o = opaque_g(W2b);
        v8f acc = vzero8f();
        const int e0 = j0 * 512 + 16 * hlf;
#pragma unroll 4
        for (int kt = 0; kt < 16; ++kt) {
          v16bf a = load_a(&sbf[l16][0], kt * 32, hlf);
          acc = wmma_bf(a, load_bg(W2o, e0 + kt * 32), acc);
        }
        const float wk = wgt[rk];
#pragma unroll
        for (int r = 0; r < 8; ++r) {
          int m = r + 8 * hlf;
          float kv = acc[r] + b2s[j0];
          kcur[m][j0] = kv;
          ksum[m][j0] = (rk == 0) ? kv : (ksum[m][j0] + wk * kv);
        }
      }
      __syncthreads();
    }

    // ---- stage 5: h_ode = LN(h + d/6 * ksum); h1 = (t==0) ? h : h_ode ----
    {
      const int m = wv, c = lane * 4;
      const float sc = dts_s[m] * (1.0f / 6.0f);
      float v[4], s = 0.f, q = 0.f;
#pragma unroll
      for (int j = 0; j < 4; ++j) {
        v[j] = hf[m][c + j] + sc * ksum[m][c + j];
        s += v[j];
        q += v[j] * v[j];
      }
      float mean, rstd;
      wave_ln_stats(s, q, mean, rstd);
#pragma unroll
      for (int j = 0; j < 4; ++j) {
        int cc = c + j;
        float hn = (v[j] - mean) * rstd * gos[cc] + bos[cc];
        float h1v = (t == 0) ? hf[m][cc] : hn;
        h1f[m][cc] = h1v;
        gruA[m][128 + cc] = f2bf(h1v);
      }
    }
    __syncthreads();

    // ---- stage 6: GRU pre-gates ----
    {  // r,z fused: [xi|h1] @ Wg[0:256]^T + (b_ih+b_hh)   (M16,N256,K256)
      const int n0 = wv * 16 + l16;
      GBF* Wgo = opaque_g(Wgb);
      v8f acc = vzero8f();
      const int e0 = n0 * 256 + 16 * hlf;
#pragma unroll 4
      for (int kt = 0; kt < 8; ++kt) {
        v16bf a = load_a(&gruA[l16][0], kt * 32, hlf);
        acc = wmma_bf(a, load_bg(Wgo, e0 + kt * 32), acc);
      }
#pragma unroll
      for (int r = 0; r < 8; ++r) {
        int m = r + 8 * hlf;
        GG_S(m, n0) = acc[r] + bsum[n0];
      }
    }
    {  // n gate halves: waves 0-7 -> i_n = xi@W_ih_n^T; waves 8-15 -> h_n = h1@W_hh_n^T
      const int wl = wv & 7;
      const int jn = wl * 16 + l16;       // 0..127
      const int nrow = 256 + jn;          // gate row in Wg
      const int koff = (wv < 8) ? 0 : 128;
      GBF* Wgo = opaque_g(Wgb);
      v8f acc = vzero8f();
      const int e0 = nrow * 256 + koff + 16 * hlf;
#pragma unroll 4
      for (int kt = 0; kt < 4; ++kt) {
        v16bf a = load_a(&gruA[l16][0], koff + kt * 32, hlf);
        acc = wmma_bf(a, load_bg(Wgo, e0 + kt * 32), acc);
      }
      if (wv < 8) {
#pragma unroll
        for (int r = 0; r < 8; ++r) {
          int m = r + 8 * hlf;
          INB_S(m, jn) = acc[r] + binn[jn];
        }
      } else {
#pragma unroll
        for (int r = 0; r < 8; ++r) {
          int m = r + 8 * hlf;
          HNB_S(m, jn) = acc[r] + bhnn[jn];
        }
      }
    }
    __syncthreads();

    // ---- stage 7: gates + LN(g_gru) -> h2; update hf; write out ----
    {
      const int m = wv, c = lane * 4;
      float v[4], s = 0.f, q = 0.f;
#pragma unroll
      for (int j = 0; j < 4; ++j) {
        int cc = c + j;
        float rg = 1.0f / (1.0f + __expf(-GG_S(m, cc)));
        float zg = 1.0f / (1.0f + __expf(-GG_S(m, 128 + cc)));
        float nn = tanhf(INB_S(m, cc) + rg * HNB_S(m, cc));
        v[j] = (1.0f - zg) * nn + zg * h1f[m][cc];
        s += v[j];
        q += v[j] * v[j];
      }
      float mean, rstd;
      wave_ln_stats(s, q, mean, rstd);
      float o[4];
#pragma unroll
      for (int j = 0; j < 4; ++j) {
        int cc = c + j;
        float h2 = (v[j] - mean) * rstd * ggs[cc] + bgs[cc];
        hf[m][cc] = h2;
        o[j] = h2;
      }
      *(float4*)(out + ((size_t)(b0 + m) * TT + t) * HH + c) =
          make_float4(o[0], o[1], o[2], o[3]);
    }
    // next-iteration stage-1/2 barriers fence reuse of fbuf / hf
  }
#undef BASE_S
#undef GG_S
#undef INB_S
#undef HNB_S
}

extern "C" void kernel_launch(void* const* d_in, const int* in_sizes, int n_in,
                              void* d_out, int out_size, void* d_ws, size_t ws_size,
                              hipStream_t stream) {
  (void)in_sizes; (void)n_in; (void)out_size; (void)ws_size;
  const float* x        = (const float*)d_in[0];
  const float* ts       = (const float*)d_in[1];
  const float* W1       = (const float*)d_in[2];
  const float* b1       = (const float*)d_in[3];
  const float* W2       = (const float*)d_in[4];
  const float* b2       = (const float*)d_in[5];
  const float* W_ih     = (const float*)d_in[6];
  const float* W_hh     = (const float*)d_in[7];
  const float* b_ih     = (const float*)d_in[8];
  const float* b_hh     = (const float*)d_in[9];
  const float* g_ode    = (const float*)d_in[10];
  const float* beta_ode = (const float*)d_in[11];
  const float* g_gru    = (const float*)d_in[12];
  const float* beta_gru = (const float*)d_in[13];

  bf16* W1b = (bf16*)d_ws;                 // 512*512 bf16
  bf16* W2b = W1b + 512 * 512;             // 128*512 bf16
  bf16* Wgb = W2b + 128 * 512;             // 384*256 bf16

  jumpode_prep_weights<<<1664, 256, 0, stream>>>(W1, W2, W_ih, W_hh,
                                                 (unsigned short*)d_ws);
  jumpode_scan<<<32, 512, 0, stream>>>(x, ts, b1, b2, b_ih, b_hh,
                                       g_ode, beta_ode, g_gru, beta_gru,
                                       W1b, W2b, Wgb, (float*)d_out);
}
